// HGTModel_48447231099383
// MI455X (gfx1250) — compile-verified, matrix operand order checked by reading, not run on hardware
//
#include <hip/hip_runtime.h>
#include <hip/hip_bf16.h>
#include <math.h>

#define C   128
#define H   4
#define Dh  32

typedef __attribute__((ext_vector_type(16))) __bf16 v16bf;
typedef __attribute__((ext_vector_type(8)))  __bf16 v8bf;
typedef __attribute__((ext_vector_type(8)))  float  v8f;

// ---------------------------------------------------------------------------
// helpers
// ---------------------------------------------------------------------------
__device__ __forceinline__ v16bf load_frag16(const __bf16* p) {
    // fragment = 8 contiguous bf16 at p (K = kb..kb+7) and 8 at p+16 (K+16..K+23)
    v8bf lo = *(const v8bf*)(p);
    v8bf hi = *(const v8bf*)(p + 16);
    v16bf r;
#pragma unroll
    for (int i = 0; i < 8; ++i) { r[i] = lo[i]; r[i + 8] = hi[i]; }
    return r;
}

__device__ __forceinline__ v16bf frag_from_f32(const float* p) {
    // same K layout, source is fp32 (LDS); convert at fragment build
    v16bf r;
#pragma unroll
    for (int i = 0; i < 8; ++i) {
        r[i]     = (__bf16)p[i];
        r[i + 8] = (__bf16)p[i + 16];
    }
    return r;
}

__device__ __forceinline__ void atomicMaxFloat(float* addr, float val) {
    // ordered-int trick: works with init = -inf
    if (val >= 0.0f) atomicMax((int*)addr, __float_as_int(val));
    else             atomicMin((unsigned int*)addr, __float_as_uint(val));
}

// ---------------------------------------------------------------------------
// weight prep: fold a_rel/m_rel into Wk/Wv, transpose all weights to [co][ci],
// convert to bf16; build effective k/v biases.
// Wt layout: [0]=Wq^T  [1]=(Wk@Arel)^T  [2]=(Wv@Mrel)^T  [3]=Wa^T
// ---------------------------------------------------------------------------
__global__ void prep_weights_kernel(const float* __restrict__ Wk,
                                    const float* __restrict__ Wq,
                                    const float* __restrict__ Wv,
                                    const float* __restrict__ Wa,
                                    const float* __restrict__ a_rel,  // [H][Dh][Dh]
                                    const float* __restrict__ m_rel,
                                    const float* __restrict__ bk,
                                    const float* __restrict__ bv,
                                    __bf16* __restrict__ Wt,          // [4][C][C]
                                    float* __restrict__ bk_eff,
                                    float* __restrict__ bv_eff) {
    int i = blockIdx.x * blockDim.x + threadIdx.x;
    if (i >= C * C) return;
    int co = i >> 7, ci = i & 127;
    int h = co >> 5, e = co & 31;

    Wt[0 * C * C + co * C + ci] = (__bf16)Wq[ci * C + co];
    float sk = 0.f, sv = 0.f;
#pragma unroll
    for (int d = 0; d < Dh; ++d) {
        float ar = a_rel[(h * Dh + d) * Dh + e];
        float mr = m_rel[(h * Dh + d) * Dh + e];
        sk += Wk[ci * C + h * Dh + d] * ar;
        sv += Wv[ci * C + h * Dh + d] * mr;
    }
    Wt[1 * C * C + co * C + ci] = (__bf16)sk;
    Wt[2 * C * C + co * C + ci] = (__bf16)sv;
    Wt[3 * C * C + co * C + ci] = (__bf16)Wa[ci * C + co];

    if (ci == 0) {
        float bks = 0.f, bvs = 0.f;
#pragma unroll
        for (int d = 0; d < Dh; ++d) {
            bks += bk[h * Dh + d] * a_rel[(h * Dh + d) * Dh + e];
            bvs += bv[h * Dh + d] * m_rel[(h * Dh + d) * Dh + e];
        }
        bk_eff[co] = bks;
        bv_eff[co] = bvs;
    }
}

// ---------------------------------------------------------------------------
// fused q/k/v GEMM: 12 waves = {q,k,v} x 4 M-subtiles of 16 rows.
// x tile staged into LDS as raw fp32 by GLOBAL_LOAD_ASYNC_TO_LDS_B128
// (ASYNCcnt path), converted to bf16 at fragment build.
// Weights pre-transposed [co][ci] -> B fragments are contiguous b128 loads.
// Inner loop: batch-load 8 B-fragments per k-slice, then 8 WMMAs.
// ---------------------------------------------------------------------------
#define XPAD 4   // fp32 row stride C+4 -> bank advance 4/row, conflict-free b128

__global__ __launch_bounds__(384)
void qkv_gemm_kernel(const float* __restrict__ x,
                     const __bf16* __restrict__ Wt,     // [3][C][C] (q,k_eff,v_eff)
                     const float* __restrict__ bq,
                     const float* __restrict__ bk_eff,
                     const float* __restrict__ bv_eff,
                     float* __restrict__ qo,
                     float* __restrict__ ko,
                     float* __restrict__ vo,
                     int n) {
    __shared__ __align__(16) float xs[64][C + XPAD];
    const int tid = threadIdx.x;
    const int rowbase = blockIdx.x * 64;

    // ---- async stage: 64x128 fp32 tile, 16B per lane-op ----
    {
        const unsigned lds_base = (unsigned)(uintptr_t)(&xs[0][0]);
        for (int i = tid; i < 64 * 32; i += 384) {     // 2048 x 16B chunks
            int r = i >> 5, seg = i & 31;
            int gr = rowbase + r;
            if (gr < n) {
                unsigned lds = lds_base + (unsigned)((r * (C + XPAD) + seg * 4) * 4);
                const float* gp = x + (size_t)gr * C + seg * 4;
                asm volatile("global_load_async_to_lds_b128 %0, %1, off"
                             :: "v"(lds), "v"(gp) : "memory");
            }
        }
        asm volatile("s_wait_asynccnt 0x0" ::: "memory");
    }
    __syncthreads();

    const int wave = tid >> 5;
    const int lane = tid & 31;
    const int wsel = wave % 3;            // 0=q 1=k 2=v
    const int mt   = wave / 3;            // 0..3
    const int lcol = lane & 15;
    const int kb   = (lane >> 4) << 3;    // 0 or 8 (ISA 16-bit A/B layout)

    const float* bias = (wsel == 0) ? bq : (wsel == 1) ? bk_eff : bv_eff;
    float*       outp = (wsel == 0) ? qo : (wsel == 1) ? ko : vo;
    const __bf16* Wsel = Wt + (size_t)wsel * C * C;

    // preload all A fragments (one DS clause), convert fp32->bf16
    v16bf afrag[4];
#pragma unroll
    for (int t = 0; t < 4; ++t)
        afrag[t] = frag_from_f32(&xs[mt * 16 + lcol][t * 32 + kb]);

    v8f acc[8];
#pragma unroll
    for (int nt = 0; nt < 8; ++nt) {
        float b = bias[nt * 16 + lcol];
#pragma unroll
        for (int r = 0; r < 8; ++r) acc[nt][r] = b;
    }

#pragma unroll
    for (int t = 0; t < 4; ++t) {
        v16bf bfr[8];
#pragma unroll
        for (int nt = 0; nt < 8; ++nt)      // 16-load clause
            bfr[nt] = load_frag16(Wsel + (size_t)(nt * 16 + lcol) * C + t * 32 + kb);
#pragma unroll
        for (int nt = 0; nt < 8; ++nt)      // 8 WMMAs, incremental waits
            acc[nt] = __builtin_amdgcn_wmma_f32_16x16x32_bf16(
                false, afrag[t], false, bfr[nt], (short)0, acc[nt], false, false);
    }

    const int rhalf = (lane >> 4) << 3;   // D rows: lanes 0-15 -> M 0..7, 16-31 -> M 8..15
#pragma unroll
    for (int nt = 0; nt < 8; ++nt) {
#pragma unroll
        for (int r = 0; r < 8; ++r) {
            int grow = rowbase + mt * 16 + rhalf + r;
            if (grow < n) outp[(size_t)grow * C + nt * 16 + lcol] = acc[nt][r];
        }
    }
}

// ---------------------------------------------------------------------------
// attention-output GEMM: h = g * (gelu(agg) @ Wa + ba) + (1-g) * x_in
// gelu fused into LDS staging (VALU transform required -> regular staging),
// skip fused into store. 8 waves, 128 rows/block.
// ---------------------------------------------------------------------------
__global__ __launch_bounds__(256)
void attnout_gemm_kernel(const float* __restrict__ agg,
                         const float* __restrict__ xin,
                         const __bf16* __restrict__ Wt,   // [C][C] Wa^T
                         const float* __restrict__ ba,
                         const float* __restrict__ skip,  // &skip[l]
                         float* __restrict__ hout,
                         int n) {
    __shared__ __align__(16) __bf16 xs[128][C + 8];
    const int tid = threadIdx.x;
    const int rowbase = blockIdx.x * 128;
    const float g = 1.f / (1.f + __expf(-skip[0]));

    for (int i = tid; i < 128 * C; i += 256) {
        int r = i >> 7, c = i & 127;
        int gr = rowbase + r;
        float v = 0.f;
        if (gr < n) {
            float a = agg[(size_t)gr * C + c];
            v = 0.5f * a * (1.f + erff(a * 0.70710678118654752f));  // exact gelu
        }
        xs[r][c] = (__bf16)v;
    }
    __syncthreads();

    const int wave = tid >> 5;            // 0..7 = M-subtile
    const int lane = tid & 31;
    const int lcol = lane & 15;
    const int kb   = (lane >> 4) << 3;

    v16bf afrag[4];
#pragma unroll
    for (int t = 0; t < 4; ++t)
        afrag[t] = load_frag16(&xs[wave * 16 + lcol][t * 32 + kb]);

    v8f acc[8];
#pragma unroll
    for (int nt = 0; nt < 8; ++nt) {
        float b = ba[nt * 16 + lcol];
#pragma unroll
        for (int r = 0; r < 8; ++r) acc[nt][r] = b;
    }

#pragma unroll
    for (int t = 0; t < 4; ++t) {
        v16bf bfr[8];
#pragma unroll
        for (int nt = 0; nt < 8; ++nt)
            bfr[nt] = load_frag16(Wt + (size_t)(nt * 16 + lcol) * C + t * 32 + kb);
#pragma unroll
        for (int nt = 0; nt < 8; ++nt)
            acc[nt] = __builtin_amdgcn_wmma_f32_16x16x32_bf16(
                false, afrag[t], false, bfr[nt], (short)0, acc[nt], false, false);
    }

    const int rhalf = (lane >> 4) << 3;
#pragma unroll
    for (int nt = 0; nt < 8; ++nt) {
#pragma unroll
        for (int r = 0; r < 8; ++r) {
            int grow = rowbase + wave * 16 + rhalf + r;
            if (grow < n) {
                size_t idx = (size_t)grow * C + nt * 16 + lcol;
                hout[idx] = g * acc[nt][r] + (1.f - g) * xin[idx];
            }
        }
    }
}

// ---------------------------------------------------------------------------
// init kernels
// ---------------------------------------------------------------------------
__global__ void init_softmax_kernel(float* __restrict__ amax,
                                    float* __restrict__ denom, int nh) {
    int i = blockIdx.x * blockDim.x + threadIdx.x;
    if (i < nh) { amax[i] = -INFINITY; denom[i] = 0.f; }
}

__global__ void zero_kernel(float4* __restrict__ p, int n4) {
    int i = blockIdx.x * blockDim.x + threadIdx.x;
    if (i < n4) p[i] = make_float4(0.f, 0.f, 0.f, 0.f);
}

// ---------------------------------------------------------------------------
// per-edge attention logits: wave per edge, float4 coalesced gathers.
// lane l covers channels 4l..4l+3 (head = l>>3); 3-step shfl_xor reduce per head.
// ---------------------------------------------------------------------------
__global__ __launch_bounds__(256)
void edge_alpha_kernel(const float* __restrict__ q,
                       const float* __restrict__ k,
                       const long long* __restrict__ src,
                       const long long* __restrict__ dst,
                       const float* __restrict__ p_rel,   // [H]
                       float* __restrict__ alpha,         // [E][H]
                       float* __restrict__ amax,          // [N][H]
                       int E) {
    int e = (int)(((size_t)blockIdx.x * 256 + threadIdx.x) >> 5);
    int lane = threadIdx.x & 31;
    if (e >= E) return;                                   // wave-uniform
    if (lane == 0 && e + 4096 < E) {                      // stream-ahead on edge list
        __builtin_prefetch(src + e + 4096, 0, 3);
        __builtin_prefetch(dst + e + 4096, 0, 3);
    }
    long long s = src[e], d = dst[e];
    float4 qv = *(const float4*)(q + (size_t)d * C + lane * 4);
    float4 kv = *(const float4*)(k + (size_t)s * C + lane * 4);
    float p = qv.x * kv.x + qv.y * kv.y + qv.z * kv.z + qv.w * kv.w;
    p += __shfl_xor(p, 1, 32);
    p += __shfl_xor(p, 2, 32);
    p += __shfl_xor(p, 4, 32);                            // sum over 8 lanes = one head
    if ((lane & 7) == 0) {
        int h = lane >> 3;
        float a = p * p_rel[h] * 0.17677669529663687f;    // 1/sqrt(32)
        alpha[(size_t)e * H + h] = a;
        atomicMaxFloat(&amax[(size_t)d * H + h], a);
    }
}

// ez = exp(alpha - amax[dst]); denom[dst] += ez   (ez written in-place on alpha)
__global__ void edge_exp_kernel(const long long* __restrict__ dst,
                                const float* __restrict__ amax,
                                float* __restrict__ ez,   // alias of alpha
                                float* __restrict__ denom,
                                int EH) {
    int i = blockIdx.x * blockDim.x + threadIdx.x;
    if (i >= EH) return;
    int e = i >> 2, h = i & 3;
    long long d = dst[e];
    float v = __expf(ez[i] - amax[(size_t)d * H + h]);
    ez[i] = v;
    atomicAdd(&denom[(size_t)d * H + h], v);
}

// agg[dst] += v[src] * attn : wave per edge, hw f32 atomics (agg fits in L2)
__global__ __launch_bounds__(256)
void edge_agg_kernel(const float* __restrict__ v,
                     const long long* __restrict__ src,
                     const long long* __restrict__ dst,
                     const float* __restrict__ ez,
                     const float* __restrict__ denom,
                     float* __restrict__ agg,
                     int E) {
    int e = (int)(((size_t)blockIdx.x * 256 + threadIdx.x) >> 5);
    int lane = threadIdx.x & 31;
    if (e >= E) return;
    if (lane == 0 && e + 4096 < E) {
        __builtin_prefetch(src + e + 4096, 0, 3);
        __builtin_prefetch(dst + e + 4096, 0, 3);
    }
    long long s = src[e], d = dst[e];
    int h = lane >> 3;
    float attn = ez[(size_t)e * H + h] / denom[(size_t)d * H + h];
    float4 vv = *(const float4*)(v + (size_t)s * C + lane * 4);
    float* ap = agg + (size_t)d * C + lane * 4;
    atomicAdd(ap + 0, vv.x * attn);
    atomicAdd(ap + 1, vv.y * attn);
    atomicAdd(ap + 2, vv.z * attn);
    atomicAdd(ap + 3, vv.w * attn);
}

// final FC: [N,128] @ [128,2] + b
__global__ void fc_kernel(const float* __restrict__ h,
                          const float* __restrict__ Wfc,
                          const float* __restrict__ bfc,
                          float* __restrict__ out, int n) {
    int i = blockIdx.x * blockDim.x + threadIdx.x;
    if (i >= n) return;
    float a0 = bfc[0], a1 = bfc[1];
    const float* hp = h + (size_t)i * C;
#pragma unroll 8
    for (int c = 0; c < C; ++c) {
        float hv = hp[c];
        a0 += hv * Wfc[c * 2 + 0];
        a1 += hv * Wfc[c * 2 + 1];
    }
    out[i * 2 + 0] = a0;
    out[i * 2 + 1] = a1;
}

// ---------------------------------------------------------------------------
// driver
// ---------------------------------------------------------------------------
extern "C" void kernel_launch(void* const* d_in, const int* in_sizes, int n_in,
                              void* d_out, int out_size, void* d_ws, size_t ws_size,
                              hipStream_t stream) {
    const float*     x      = (const float*)d_in[0];
    const long long* eidx   = (const long long*)d_in[1];   // int64 per reference
    const float*     Wk     = (const float*)d_in[2];
    const float*     bk     = (const float*)d_in[3];
    const float*     Wq     = (const float*)d_in[4];
    const float*     bq     = (const float*)d_in[5];
    const float*     Wv     = (const float*)d_in[6];
    const float*     bv     = (const float*)d_in[7];
    const float*     a_rel  = (const float*)d_in[8];
    const float*     m_rel  = (const float*)d_in[9];
    const float*     p_rel  = (const float*)d_in[10];
    const float*     Wa     = (const float*)d_in[11];
    const float*     ba     = (const float*)d_in[12];
    const float*     skip   = (const float*)d_in[13];
    const float*     Wfc    = (const float*)d_in[14];
    const float*     bfc    = (const float*)d_in[15];
    float*           out    = (float*)d_out;

    const int N = in_sizes[0] / C;
    const int E = in_sizes[1] / 2;
    const long long* src = eidx;
    const long long* dst = eidx + E;

    const size_t NC = (size_t)N * C;
    float* ws    = (float*)d_ws;
    float* buf0  = ws;                       // q, later agg
    float* buf1  = buf0 + NC;                // k, later h2
    float* buf2  = buf1 + NC;                // v
    float* hbuf  = buf2 + NC;                // h after layer 0
    float* alpha = hbuf + NC;                // [E][H], reused as ez
    float* amax  = alpha + (size_t)E * H;
    float* denom = amax + (size_t)N * H;
    float* beff  = denom + (size_t)N * H;    // 2 layers * {bk_eff, bv_eff} * C
    __bf16* Wt   = (__bf16*)(beff + 4 * C);  // 2 layers * 4 * C * C bf16

    const int gPrep = (C * C + 255) / 256;
    const int gQKV  = (N + 63) / 64;
    const int gOut  = (N + 127) / 128;
    const int gNH   = (N * H + 255) / 256;
    const int gNC4  = (int)((NC / 4 + 255) / 256);
    const int gEw   = (E + 7) / 8;           // wave-per-edge kernels
    const int gEH   = (E * H + 255) / 256;
    const int gFC   = (N + 255) / 256;

    // weight prep for both layers
    for (int l = 0; l < 2; ++l) {
        prep_weights_kernel<<<gPrep, 256, 0, stream>>>(
            Wk + (size_t)l * C * C, Wq + (size_t)l * C * C,
            Wv + (size_t)l * C * C, Wa + (size_t)l * C * C,
            a_rel + (size_t)l * H * Dh * Dh, m_rel + (size_t)l * H * Dh * Dh,
            bk + (size_t)l * C, bv + (size_t)l * C,
            Wt + (size_t)l * 4 * C * C,
            beff + (size_t)l * 2 * C, beff + (size_t)l * 2 * C + C);
    }

    for (int l = 0; l < 2; ++l) {
        const float* xin  = (l == 0) ? x : hbuf;
        float*       hout = (l == 0) ? hbuf : buf1;  // buf1 (k) is dead by then
        const __bf16* Wtl = Wt + (size_t)l * 4 * C * C;
        const float* bkeff = beff + (size_t)l * 2 * C;
        const float* bveff = bkeff + C;

        // q,k,v = x @ {Wq, Wk@Arel, Wv@Mrel} + bias   (WMMA bf16, async LDS stage)
        qkv_gemm_kernel<<<gQKV, 384, 0, stream>>>(
            xin, Wtl, bq + (size_t)l * C, bkeff, bveff, buf0, buf1, buf2, N);

        // segment softmax
        init_softmax_kernel<<<gNH, 256, 0, stream>>>(amax, denom, N * H);
        edge_alpha_kernel<<<gEw, 256, 0, stream>>>(
            buf0, buf1, src, dst, p_rel + (size_t)l * H, alpha, amax, E);
        edge_exp_kernel<<<gEH, 256, 0, stream>>>(dst, amax, alpha, denom, E * H);

        // message aggregation (agg reuses q's buffer)
        zero_kernel<<<gNC4, 256, 0, stream>>>((float4*)buf0, (int)(NC / 4));
        edge_agg_kernel<<<gEw, 256, 0, stream>>>(buf2, src, dst, alpha, denom, buf0, E);

        // h = sigmoid(skip) * (gelu(agg)@Wa + ba) + (1-sigmoid(skip)) * x
        attnout_gemm_kernel<<<gOut, 256, 0, stream>>>(
            buf0, xin, Wtl + 3 * C * C, ba + (size_t)l * C, skip + l, hout, N);
    }

    fc_kernel<<<gFC, 256, 0, stream>>>(buf1, Wfc, bfc, out, N);
}
